// BatchedMoE_7017976561989
// MI455X (gfx1250) — matrix operand order repaired and checked by reading, not compiled, not run on hardware
//
#include <hip/hip_runtime.h>
#include <math.h>

#define NTOK  8192
#define CDIM  1024
#define NEXP  8
#define HDIM  1408
#define HSDIM 2816
#define TILE_M 128
#define TILE_N 128
#define TILE_K 64
#define CAP   (NTOK*2 + NEXP*TILE_M)   // 17408, fixed padded row capacity

#define USE_ASYNC 1   // global_load_async_to_lds_b128 staging (ASYNCcnt-tracked)

typedef __attribute__((ext_vector_type(16))) __bf16 v16bf;
typedef __attribute__((ext_vector_type(8)))  float  v8f;

union AFrag { v16bf v; uint4 q[2]; };

__device__ __forceinline__ unsigned short f2bf(float f) {
  unsigned u = __float_as_uint(f);
  u += 0x7FFFu + ((u >> 16) & 1u);          // round-to-nearest-even
  return (unsigned short)(u >> 16);
}
__device__ __forceinline__ unsigned pack2bf(float lo, float hi) {
  return (unsigned)f2bf(lo) | ((unsigned)f2bf(hi) << 16);
}
__device__ __forceinline__ v8f wmma_bf16(const AFrag& a, const AFrag& b, v8f c) {
  return __builtin_amdgcn_wmma_f32_16x16x32_bf16(false, a.v, false, b.v, (short)0, c, false, false);
}

// 16-byte global -> LDS copy. Async path: per-lane VDST = LDS byte address
// (low 32 bits of the generic shared pointer), VADDR = 64-bit global address.
__device__ __forceinline__ void cp16(unsigned short* lds, const unsigned short* g) {
#if USE_ASYNC
  asm volatile("global_load_async_to_lds_b128 %0, %1, off"
               :: "v"((unsigned)(size_t)lds),
                  "v"((unsigned long long)(size_t)g)
               : "memory");
#else
  *(uint4*)lds = *(const uint4*)g;
#endif
}
__device__ __forceinline__ void cp_wait() {
#if USE_ASYNC
  asm volatile("s_wait_asynccnt 0" ::: "memory");
#endif
}

// ---------------------------------------------------------------- router ----
__global__ __launch_bounds__(256)
void moe_router(const float* __restrict__ x, const float* __restrict__ Wg,
                int* __restrict__ idxO, float* __restrict__ probO, int* __restrict__ counts) {
  const int wave = threadIdx.x >> 5, lane = threadIdx.x & 31;
  const int t = blockIdx.x * 8 + wave;
  const float* xr = x + (size_t)t * CDIM;
  float acc[NEXP];
#pragma unroll
  for (int e = 0; e < NEXP; ++e) acc[e] = 0.f;
  for (int c = lane; c < CDIM; c += 32) {
    float xv = xr[c];
    const float4* w = (const float4*)(Wg + (size_t)c * NEXP);
    float4 w0 = w[0], w1 = w[1];
    acc[0] += xv * w0.x; acc[1] += xv * w0.y; acc[2] += xv * w0.z; acc[3] += xv * w0.w;
    acc[4] += xv * w1.x; acc[5] += xv * w1.y; acc[6] += xv * w1.z; acc[7] += xv * w1.w;
  }
#pragma unroll
  for (int off = 16; off >= 1; off >>= 1)
#pragma unroll
    for (int e = 0; e < NEXP; ++e) acc[e] += __shfl_xor(acc[e], off, 32);
  if (lane == 0) {
    int e0 = 0; float l0 = acc[0];
#pragma unroll
    for (int e = 1; e < NEXP; ++e) if (acc[e] > l0) { l0 = acc[e]; e0 = e; }
    int e1 = (e0 == 0) ? 1 : 0; float l1 = acc[e1];
#pragma unroll
    for (int e = 0; e < NEXP; ++e) if (e != e0 && acc[e] > l1) { l1 = acc[e]; e1 = e; }
    float m = fmaxf(l0, l1);
    float p0 = __expf(l0 - m), p1 = __expf(l1 - m);
    float s = p0 + p1;
    idxO[2 * t] = e0;      idxO[2 * t + 1] = e1;
    probO[2 * t] = p0 / s; probO[2 * t + 1] = p1 / s;
    atomicAdd(&counts[e0], 1);
    atomicAdd(&counts[e1], 1);
  }
}

// -------------------------------------------------------- offsets + scatter -
__global__ void moe_offsets(const int* __restrict__ counts, int* __restrict__ offs,
                            int* __restrict__ cursor) {
  if (threadIdx.x == 0) {
    int o = 0; offs[0] = 0;
    for (int e = 0; e < NEXP; ++e) {
      cursor[e] = o;
      o += (counts[e] + TILE_M - 1) & ~(TILE_M - 1);   // pad segment to tile M
      offs[e + 1] = o;
    }
  }
}

__global__ __launch_bounds__(256)
void moe_scatter(const int* __restrict__ idx, const float* __restrict__ prob,
                 int* __restrict__ cursor, int* __restrict__ perm, float* __restrict__ rowp) {
  int t = blockIdx.x * 256 + threadIdx.x;
#pragma unroll
  for (int j = 0; j < 2; ++j) {
    int e = idx[2 * t + j];
    int r = atomicAdd(&cursor[e], 1);
    perm[r] = t;
    rowp[r] = prob[2 * t + j];
  }
}

// -------------------------------------------- one-time precision conversion -
// x fp32 [NTOK][C] -> bf16 (same layout)
__global__ __launch_bounds__(256)
void xconvert(const float* __restrict__ x, unsigned short* __restrict__ xb) {
  size_t i = ((size_t)blockIdx.x * 256 + threadIdx.x) * 4;
  float4 f = *(const float4*)(x + i);
  unsigned* d = (unsigned*)(xb + i);
  d[0] = pack2bf(f.x, f.y);
  d[1] = pack2bf(f.z, f.w);
}

// W fp32 [G][Kd][Nd] -> bf16 [G][Nd][Kd] (convert + transpose, 64x64 tiles)
__global__ __launch_bounds__(256)
void wconvert(const float* __restrict__ src, unsigned short* __restrict__ dst,
              int Kd, int Nd) {
  __shared__ unsigned short lT[64][65];
  const size_t matS = (size_t)Kd * Nd;
  const float* s = src + (size_t)blockIdx.z * matS;
  unsigned short* d = dst + (size_t)blockIdx.z * matS;
  const int n0 = blockIdx.x * 64, k0 = blockIdx.y * 64;
  const int tid = threadIdx.x;
  {
    const int kk = tid >> 2;
    const int nc = (tid & 3) * 16;
    const float4* p = (const float4*)(s + (size_t)(k0 + kk) * Nd + n0 + nc);
#pragma unroll
    for (int i = 0; i < 4; ++i) {
      float4 f = p[i];
      lT[kk][nc + i * 4 + 0] = f2bf(f.x);
      lT[kk][nc + i * 4 + 1] = f2bf(f.y);
      lT[kk][nc + i * 4 + 2] = f2bf(f.z);
      lT[kk][nc + i * 4 + 3] = f2bf(f.w);
    }
  }
  __syncthreads();
  {
    const int nn = tid >> 2;
    const int kc = (tid & 3) * 16;
    unsigned* q = (unsigned*)(d + (size_t)(n0 + nn) * Kd + k0 + kc);
#pragma unroll
    for (int i = 0; i < 8; ++i)
      q[i] = (unsigned)lT[kc + 2 * i][nn] | ((unsigned)lT[kc + 2 * i + 1][nn] << 16);
  }
}

// ------------------------------------------------------------ LDS staging ---
// Copy a 128-row x 64-k bf16 tile from row-major [rows][ldK] source into LDS.
__device__ __forceinline__ void stage_rows(unsigned short* lds, const unsigned short* src,
                                           size_t ldK, int k0, int rowBase, int tid) {
  const int r = tid >> 1;               // 0..127
  const int kc = (tid & 1) * 32;        // 0 or 32
  const unsigned short* s = src + (size_t)(rowBase + r) * ldK + k0 + kc;
  unsigned short* d = lds + r * TILE_K + kc;
#pragma unroll
  for (int i = 0; i < 4; ++i) cp16(d + i * 8, s + i * 8);
  __builtin_prefetch(s + TILE_K, 0, 1);   // next K strip -> global_prefetch_b8
}

__device__ __forceinline__ void load_afrag(AFrag& f, const unsigned short* lA, int m, int kb) {
  f.q[0] = *(const uint4*)&lA[m * TILE_K + kb];
  f.q[1] = *(const uint4*)&lA[m * TILE_K + kb + 8];
}
__device__ __forceinline__ void load_bfrag(AFrag& f, const unsigned short* lB, int n, int kh) {
  f.q[0] = *(const uint4*)&lB[n * TILE_K + kh];
  f.q[1] = *(const uint4*)&lB[n * TILE_K + kh + 8];
}

// ----------------------------------------------- fused fc1/fc2 (silu * up) --
// Hout[row, nCols] = silu(X W1t^T) * (X W2t^T); W*t are bf16 [nCols][CDIM].
template <bool ROUTED>
__global__ __launch_bounds__(256)
void moe_fc(const unsigned short* __restrict__ XB, const unsigned short* __restrict__ W1t,
            const unsigned short* __restrict__ W2t, const int* __restrict__ perm,
            const int* __restrict__ offs, unsigned short* __restrict__ Hout, int nCols) {
  __shared__ unsigned short lA[TILE_M * TILE_K];
  __shared__ unsigned short lB1[TILE_N * TILE_K];
  __shared__ unsigned short lB2[TILE_N * TILE_K];
  __shared__ int sTok[TILE_M];

  const int tid = threadIdx.x;
  const int wave = tid >> 5, lane = tid & 31;
  const int rowStart = blockIdx.y * TILE_M;
  const int colStart = blockIdx.x * TILE_N;

  const unsigned short* W1 = W1t;
  const unsigned short* W2 = W2t;
  if (ROUTED) {
    int e = 0;
#pragma unroll
    for (int i = 0; i < NEXP; ++i) if (rowStart >= offs[i + 1]) e++;
    if (e >= NEXP) e = NEXP - 1;
    W1 = W1t + (size_t)e * nCols * CDIM;
    W2 = W2t + (size_t)e * nCols * CDIM;
    if (tid < TILE_M) sTok[tid] = perm[rowStart + tid];
    __syncthreads();
  }

  v8f acc1[2][4], acc2[2][4];
#pragma unroll
  for (int mi = 0; mi < 2; ++mi)
#pragma unroll
    for (int ni = 0; ni < 4; ++ni)
#pragma unroll
      for (int i = 0; i < 8; ++i) { acc1[mi][ni][i] = 0.f; acc2[mi][ni][i] = 0.f; }

  const int mBase = (wave >> 1) * 32;     // 4 waves along M
  const int nBase = (wave & 1) * 64;      // 2 waves along N
  const int lane15 = lane & 15;
  const int hiHalf = lane >> 4;

  for (int k0 = 0; k0 < CDIM; k0 += TILE_K) {
    // ---- stage A (bf16 copy, optional gather via perm)
    {
      const int r = tid >> 1;
      const int kc = (tid & 1) * 32;
      int tok = ROUTED ? sTok[r] : (rowStart + r);
      unsigned short* d = &lA[r * TILE_K + kc];
      if (ROUTED && tok < 0) {
        uint4 z = make_uint4(0u, 0u, 0u, 0u);
#pragma unroll
        for (int i = 0; i < 4; ++i) *(uint4*)(d + i * 8) = z;
      } else {
        const unsigned short* s = XB + (size_t)tok * CDIM + k0 + kc;
#pragma unroll
        for (int i = 0; i < 4; ++i) cp16(d + i * 8, s + i * 8);
      }
    }
    // ---- stage B tiles (already bf16 [n][k] in global)
    stage_rows(lB1, W1, CDIM, k0, colStart, tid);
    stage_rows(lB2, W2, CDIM, k0, colStart, tid);
    cp_wait();
    __syncthreads();

#pragma unroll
    for (int s = 0; s < 2; ++s) {
      const int ks = s * 32;
      AFrag a[2], b1[4], b2[4];
#pragma unroll
      for (int mi = 0; mi < 2; ++mi)
        load_afrag(a[mi], lA, mBase + mi * 16 + lane15, ks + hiHalf * 16);
#pragma unroll
      for (int ni = 0; ni < 4; ++ni) {
        load_bfrag(b1[ni], lB1, nBase + ni * 16 + lane15, ks + hiHalf * 16);
        load_bfrag(b2[ni], lB2, nBase + ni * 16 + lane15, ks + hiHalf * 16);
      }
#pragma unroll
      for (int mi = 0; mi < 2; ++mi)
#pragma unroll
        for (int ni = 0; ni < 4; ++ni) {
          acc1[mi][ni] = wmma_bf16(a[mi], b1[ni], acc1[mi][ni]);
          acc2[mi][ni] = wmma_bf16(a[mi], b2[ni], acc2[mi][ni]);
        }
    }
    __syncthreads();
  }

  // ---- epilogue: h = silu(g) * u, store bf16
#pragma unroll
  for (int mi = 0; mi < 2; ++mi)
#pragma unroll
    for (int ni = 0; ni < 4; ++ni) {
      const int col = colStart + nBase + ni * 16 + lane15;
      const int row0 = rowStart + mBase + mi * 16 + hiHalf * 8;
      v8f g = acc1[mi][ni], u = acc2[mi][ni];
#pragma unroll
      for (int v = 0; v < 8; ++v) {
        float gv = g[v];
        float h = (gv / (1.f + __expf(-gv))) * u[v];
        Hout[(size_t)(row0 + v) * nCols + col] = f2bf(h);
      }
    }
}

// --------------------------------------------------------------- projection -
// ROUTED: y[tok] += prob * (H W3t^T[e]);  shared: y[row] = H Ws3t^T.
// W*t are bf16 [CDIM][Kd] (pre-transposed).
template <bool ROUTED>
__global__ __launch_bounds__(256)
void moe_proj(const unsigned short* __restrict__ Hin, const unsigned short* __restrict__ Wt,
              const int* __restrict__ perm, const float* __restrict__ rowp,
              const int* __restrict__ offs, float* __restrict__ Y, int Kd) {
  __shared__ unsigned short lA[TILE_M * TILE_K];
  __shared__ unsigned short lB[TILE_N * TILE_K];
  __shared__ int sTok[TILE_M];
  __shared__ float sP[TILE_M];

  const int tid = threadIdx.x;
  const int wave = tid >> 5, lane = tid & 31;
  const int rowStart = blockIdx.y * TILE_M;
  const int colStart = blockIdx.x * TILE_N;

  const unsigned short* W = Wt;
  if (ROUTED) {
    int e = 0;
#pragma unroll
    for (int i = 0; i < NEXP; ++i) if (rowStart >= offs[i + 1]) e++;
    if (e >= NEXP) e = NEXP - 1;
    W = Wt + (size_t)e * CDIM * Kd;
    if (tid < TILE_M) { sTok[tid] = perm[rowStart + tid]; sP[tid] = rowp[rowStart + tid]; }
    __syncthreads();
  }

  v8f acc[2][4];
#pragma unroll
  for (int mi = 0; mi < 2; ++mi)
#pragma unroll
    for (int ni = 0; ni < 4; ++ni)
#pragma unroll
      for (int i = 0; i < 8; ++i) acc[mi][ni][i] = 0.f;

  const int mBase = (wave >> 1) * 32;
  const int nBase = (wave & 1) * 64;
  const int lane15 = lane & 15;
  const int hiHalf = lane >> 4;

  for (int k0 = 0; k0 < Kd; k0 += TILE_K) {
    stage_rows(lA, Hin, Kd, k0, rowStart, tid);   // A: bf16 copy
    stage_rows(lB, W, Kd, k0, colStart, tid);     // B: bf16 copy
    cp_wait();
    __syncthreads();

#pragma unroll
    for (int s = 0; s < 2; ++s) {
      const int ks = s * 32;
      AFrag a[2], b[4];
#pragma unroll
      for (int mi = 0; mi < 2; ++mi)
        load_afrag(a[mi], lA, mBase + mi * 16 + lane15, ks + hiHalf * 16);
#pragma unroll
      for (int ni = 0; ni < 4; ++ni)
        load_bfrag(b[ni], lB, nBase + ni * 16 + lane15, ks + hiHalf * 16);
#pragma unroll
      for (int mi = 0; mi < 2; ++mi)
#pragma unroll
        for (int ni = 0; ni < 4; ++ni)
          acc[mi][ni] = wmma_bf16(a[mi], b[ni], acc[mi][ni]);
    }
    __syncthreads();
  }

#pragma unroll
  for (int mi = 0; mi < 2; ++mi)
#pragma unroll
    for (int ni = 0; ni < 4; ++ni) {
      const int col = colStart + nBase + ni * 16 + lane15;
      const int mLoc0 = mBase + mi * 16 + hiHalf * 8;
      v8f r = acc[mi][ni];
#pragma unroll
      for (int v = 0; v < 8; ++v) {
        const int mLoc = mLoc0 + v;
        if (ROUTED) {
          int tok = sTok[mLoc];
          if (tok >= 0) atomicAdd(&Y[(size_t)tok * CDIM + col], sP[mLoc] * r[v]);
        } else {
          Y[(size_t)(rowStart + mLoc) * CDIM + col] = r[v];
        }
      }
    }
}

// ----------------------------------------------------------- workspace map --
static constexpr size_t A256(size_t x) { return (x + 255) & ~(size_t)255; }
static constexpr size_t OFF_IDX  = 0;
static constexpr size_t OFF_PROB = OFF_IDX  + A256((size_t)NTOK * 2 * 4);
static constexpr size_t OFF_CNT  = OFF_PROB + A256((size_t)NTOK * 2 * 4);
static constexpr size_t OFF_OFFS = OFF_CNT  + 256;
static constexpr size_t OFF_CUR  = OFF_OFFS + 256;
static constexpr size_t OFF_PERM = OFF_CUR  + 256;
static constexpr size_t OFF_ROWP = OFF_PERM + A256((size_t)CAP * 4);
static constexpr size_t OFF_XB   = OFF_ROWP + A256((size_t)CAP * 4);
static constexpr size_t OFF_WT1  = OFF_XB   + A256((size_t)NTOK * CDIM * 2);
static constexpr size_t OFF_WT2  = OFF_WT1  + A256((size_t)NEXP * CDIM * HDIM * 2);
static constexpr size_t OFF_WT3  = OFF_WT2  + A256((size_t)NEXP * CDIM * HDIM * 2);
static constexpr size_t OFF_WS1  = OFF_WT3  + A256((size_t)NEXP * CDIM * HDIM * 2);
static constexpr size_t OFF_WS2  = OFF_WS1  + A256((size_t)CDIM * HSDIM * 2);
static constexpr size_t OFF_WS3  = OFF_WS2  + A256((size_t)CDIM * HSDIM * 2);
static constexpr size_t OFF_HH   = OFF_WS3  + A256((size_t)CDIM * HSDIM * 2);
static constexpr size_t OFF_HS   = OFF_HH   + A256((size_t)CAP * HDIM * 2);
// total ~= 199 MB

extern "C" void kernel_launch(void* const* d_in, const int* in_sizes, int n_in,
                              void* d_out, int out_size, void* d_ws, size_t ws_size,
                              hipStream_t stream) {
  (void)in_sizes; (void)n_in; (void)out_size; (void)ws_size;
  const float* x   = (const float*)d_in[0];
  const float* Wg  = (const float*)d_in[1];
  const float* W1  = (const float*)d_in[2];
  const float* W2  = (const float*)d_in[3];
  const float* W3  = (const float*)d_in[4];
  const float* Ws1 = (const float*)d_in[5];
  const float* Ws2 = (const float*)d_in[6];
  const float* Ws3 = (const float*)d_in[7];
  float* y = (float*)d_out;

  char* ws = (char*)d_ws;
  int*            idx    = (int*)   (ws + OFF_IDX);
  float*          prob   = (float*) (ws + OFF_PROB);
  int*            counts = (int*)   (ws + OFF_CNT);
  int*            offs   = (int*)   (ws + OFF_OFFS);
  int*            cursor = (int*)   (ws + OFF_CUR);
  int*            perm   = (int*)   (ws + OFF_PERM);
  float*          rowp   = (float*) (ws + OFF_ROWP);
  unsigned short* xb     = (unsigned short*)(ws + OFF_XB);
  unsigned short* Wt1    = (unsigned short*)(ws + OFF_WT1);
  unsigned short* Wt2    = (unsigned short*)(ws + OFF_WT2);
  unsigned short* Wt3    = (unsigned short*)(ws + OFF_WT3);
  unsigned short* Wts1   = (unsigned short*)(ws + OFF_WS1);
  unsigned short* Wts2   = (unsigned short*)(ws + OFF_WS2);
  unsigned short* Wts3   = (unsigned short*)(ws + OFF_WS3);
  unsigned short* Hh     = (unsigned short*)(ws + OFF_HH);
  unsigned short* Hs     = (unsigned short*)(ws + OFF_HS);

  hipMemsetAsync(counts, 0, NEXP * sizeof(int), stream);
  hipMemsetAsync(perm, 0xFF, (size_t)CAP * sizeof(int), stream);   // -1 padding rows

  moe_router <<<NTOK / 8, 256, 0, stream>>>(x, Wg, idx, prob, counts);
  moe_offsets<<<1, 32, 0, stream>>>(counts, offs, cursor);
  moe_scatter<<<NTOK / 256, 256, 0, stream>>>(idx, prob, cursor, perm, rowp);

  // one-time bf16 conversion (+ weight transpose to [N][K])
  xconvert<<<(NTOK * CDIM) / 1024, 256, 0, stream>>>(x, xb);
  wconvert<<<dim3(HDIM / 64, CDIM / 64, NEXP), 256, 0, stream>>>(W1, Wt1, CDIM, HDIM);
  wconvert<<<dim3(HDIM / 64, CDIM / 64, NEXP), 256, 0, stream>>>(W2, Wt2, CDIM, HDIM);
  wconvert<<<dim3(CDIM / 64, HDIM / 64, NEXP), 256, 0, stream>>>(W3, Wt3, HDIM, CDIM);
  wconvert<<<dim3(HSDIM / 64, CDIM / 64, 1),  256, 0, stream>>>(Ws1, Wts1, CDIM, HSDIM);
  wconvert<<<dim3(HSDIM / 64, CDIM / 64, 1),  256, 0, stream>>>(Ws2, Wts2, CDIM, HSDIM);
  wconvert<<<dim3(CDIM / 64, HSDIM / 64, 1),  256, 0, stream>>>(Ws3, Wts3, HSDIM, CDIM);

  // shared expert: h = silu(x Ws1) * (x Ws2); y = h Ws3  (initializes y)
  moe_fc<false>  <<<dim3(HSDIM / TILE_N, NTOK / TILE_M), 256, 0, stream>>>(
      xb, Wts1, Wts2, nullptr, nullptr, Hs, HSDIM);
  moe_proj<false><<<dim3(CDIM / TILE_N, NTOK / TILE_M), 256, 0, stream>>>(
      Hs, Wts3, nullptr, nullptr, nullptr, y, HSDIM);

  // routed experts (gathered rows): h = silu(xg W1[e]) * (xg W2[e]); y += p * h W3[e]
  moe_fc<true>   <<<dim3(HDIM / TILE_N, CAP / TILE_M), 256, 0, stream>>>(
      xb, Wt1, Wt2, perm, offs, Hh, HDIM);
  moe_proj<true> <<<dim3(CDIM / TILE_N, CAP / TILE_M), 256, 0, stream>>>(
      Hh, Wt3, perm, rowp, offs, y, HDIM);
}
